// QKVAttention_31181462569018
// MI455X (gfx1250) — matrix operand order repaired and checked by reading, not compiled, not run on hardware
//
#include <hip/hip_runtime.h>
#include <hip/hip_bf16.h>

typedef __attribute__((ext_vector_type(16))) _Float16 v16h;
typedef __attribute__((ext_vector_type(8)))  float    v8f;

#define HEADS  16
#define CH     64
#define SEQ    2048
#define T_TILE 128
#define S_TILE 64

// LDS layout (bytes):
//   sK : [64 s][66 h]  f16  ->  8448   (K tile, transposed, score-B gathers)
//   sV : [64 c][66 h]  f16  ->  8448   (V tile, AV-B gathers)
//   sQ : [128 t][66 h] f16  -> 16896   (Q tile, A gathers; dead after reg load)
//   sP : 8 waves x [16 t][66 h] f16 -> 16896 (per-wave P scratch)
//   out staging reuses sQ+sP region: [64 c][132 t] f32 = 33792
#define SMEM_BYTES 50688

union FragH { unsigned u[8]; v16h v; };

template <int M>
__device__ __forceinline__ float swz_xor(float x) {
    int i = __builtin_bit_cast(int, x);
    // group-of-32 swizzle: and=0x1f, or=0, xor=M  -> lane reads lane^M
    i = __builtin_amdgcn_ds_swizzle(i, (M << 10) | 0x1F);
    return __builtin_bit_cast(float, i);
}

__device__ __forceinline__ float red_max16(float x) {
    x = fmaxf(x, swz_xor<1>(x));
    x = fmaxf(x, swz_xor<2>(x));
    x = fmaxf(x, swz_xor<4>(x));
    x = fmaxf(x, swz_xor<8>(x));
    return x;
}
__device__ __forceinline__ float red_sum16(float x) {
    x += swz_xor<1>(x);
    x += swz_xor<2>(x);
    x += swz_xor<4>(x);
    x += swz_xor<8>(x);
    return x;
}

__global__ __launch_bounds__(256) void qkv_attn_kernel(const float* __restrict__ qkv,
                                                       float* __restrict__ out) {
    __shared__ __align__(16) unsigned char smem[SMEM_BYTES];
    _Float16* sK = (_Float16*)(smem + 0);
    _Float16* sV = (_Float16*)(smem + 8448);
    _Float16* sQ = (_Float16*)(smem + 16896);
    _Float16* sP = (_Float16*)(smem + 33792);
    float*  sOut = (float*)(smem + 16896);   // reuses sQ+sP after final barrier

    const int tid  = threadIdx.x;
    const int wave = tid >> 5;
    const int lane = tid & 31;
    const int l16  = lane & 15;
    const int half = lane >> 4;

    const int bh = blockIdx.y;        // 0..63 = b*16 + head
    const int b  = bh >> 4;
    const int hh = bh & 15;
    const int t0 = blockIdx.x * T_TILE;

    const size_t hs = (size_t)CH * SEQ;
    const float* qp = qkv + ((size_t)b * 3 * HEADS +             hh) * hs;
    const float* kp = qkv + ((size_t)b * 3 * HEADS + HEADS     + hh) * hs;
    const float* vp = qkv + ((size_t)b * 3 * HEADS + 2 * HEADS + hh) * hs;

    // ---- Stage Q tile (x 1/8 total softmax scale) transposed: sQ[t][c] ----
    for (int i = tid; i < T_TILE * CH; i += 256) {
        int c = i >> 7, tl = i & 127;
        sQ[tl * 66 + c] = (_Float16)(qp[(size_t)c * SEQ + t0 + tl] * 0.125f);
    }
    __syncthreads();

    // ---- Q A-fragments (16x32, 2 chunks over c) into registers, reused all steps ----
    FragH aQ[2];
    {
        const int trow = 16 * wave + l16;            // A: M = lane%16 (both halves)
        #pragma unroll
        for (int kc = 0; kc < 2; ++kc)
            #pragma unroll
            for (int v = 0; v < 8; ++v) {
                int koff = 32 * kc + 2 * v + (v >= 4 ? 8 : 0) + 8 * half;
                aQ[kc].u[v] = *(const unsigned*)(sQ + trow * 66 + koff);
            }
    }

    v8f Oacc[4];
    #pragma unroll
    for (int ct = 0; ct < 4; ++ct) Oacc[ct] = {};
    float rMax[8], rSum[8];
    #pragma unroll
    for (int r = 0; r < 8; ++r) { rMax[r] = -3.0e38f; rSum[r] = 0.0f; }

    _Float16* sPw = sP + wave * (16 * 66);

    // ---- Register double-buffer for K/V global tiles (16 elems each/thread) ----
    float kreg[16], vreg[16];
    #pragma unroll
    for (int j = 0; j < 16; ++j) {
        int idx = tid + j * 256;                 // 0..4095
        int c = idx >> 6, sl = idx & 63;
        kreg[j] = kp[(size_t)c * SEQ + sl];      // tile s0 = 0
        vreg[j] = vp[(size_t)c * SEQ + sl];
    }

    for (int s0 = 0; s0 < SEQ; s0 += S_TILE) {
        __syncthreads();                 // previous step's sK/sV consumers done
        // ---- Dump prefetched registers (f32 -> f16) into LDS tiles ----
        #pragma unroll
        for (int j = 0; j < 16; ++j) {
            int idx = tid + j * 256;
            int c = idx >> 6, sl = idx & 63;
            sK[sl * 66 + c] = (_Float16)kreg[j];
            sV[c * 66 + sl] = (_Float16)vreg[j];
        }
        __syncthreads();

        // ---- Issue next tile's global loads NOW; they complete during compute ----
        if (s0 + S_TILE < SEQ) {
            #pragma unroll
            for (int j = 0; j < 16; ++j) {
                int idx = tid + j * 256;
                int c = idx >> 6, sl = idx & 63;
                kreg[j] = kp[(size_t)c * SEQ + s0 + S_TILE + sl];
                vreg[j] = vp[(size_t)c * SEQ + s0 + S_TILE + sl];
            }
            if (s0 + 2 * S_TILE < SEQ) { // warm caches for the tile after that
                int c = tid >> 2, seg = tid & 3;
                __builtin_prefetch(kp + (size_t)c * SEQ + s0 + 2 * S_TILE + seg * 16, 0, 3);
                __builtin_prefetch(vp + (size_t)c * SEQ + s0 + 2 * S_TILE + seg * 16, 0, 3);
            }
        }

        // ---- Scores: S(16t x 64s) = Qa^T K  via 4 n-tiles x 2 k-chunks WMMA ----
        v8f Sc[4];
        #pragma unroll
        for (int nt = 0; nt < 4; ++nt) {
            v8f acc = {};
            const int srow = 16 * nt + l16;          // B: N = lane%16
            #pragma unroll
            for (int kc = 0; kc < 2; ++kc) {
                FragH bK;
                #pragma unroll
                for (int v = 0; v < 8; ++v) {
                    int coff = 32 * kc + 2 * v + (v >= 4 ? 8 : 0) + 8 * half;   // B: K = 2v + (v>=4?8:0) + 8*half
                    bK.u[v] = *(const unsigned*)(sK + srow * 66 + coff);
                }
                acc = __builtin_amdgcn_wmma_f32_16x16x32_f16(
                        false, aQ[kc].v, false, bK.v, (short)0, acc, false, false);
            }
            Sc[nt] = acc;
        }

        // ---- Online softmax: C-frag row = r + 8*half lives across 16 lanes ----
        float factor[8], psum[8];
        #pragma unroll
        for (int r = 0; r < 8; ++r) {
            float m = fmaxf(fmaxf(Sc[0][r], Sc[1][r]), fmaxf(Sc[2][r], Sc[3][r]));
            m = red_max16(m);
            float nm = fmaxf(rMax[r], m);
            factor[r] = __expf(rMax[r] - nm);
            rMax[r] = nm;
            psum[r] = 0.0f;
        }
        #pragma unroll
        for (int nt = 0; nt < 4; ++nt)
            #pragma unroll
            for (int r = 0; r < 8; ++r) {
                float p = __expf(Sc[nt][r] - rMax[r]);
                psum[r] += p;
                sPw[(r + 8 * half) * 66 + 16 * nt + l16] = (_Float16)p;
            }
        #pragma unroll
        for (int r = 0; r < 8; ++r) {
            rSum[r] = rSum[r] * factor[r] + red_sum16(psum[r]);
            #pragma unroll
            for (int ct = 0; ct < 4; ++ct) Oacc[ct][r] *= factor[r];
        }

        // ---- O(16t x 64c) += P(16 x 64) * V^T : 2 k-chunks x 4 c-tiles WMMA ----
        #pragma unroll
        for (int kc = 0; kc < 2; ++kc) {
            FragH aP;
            #pragma unroll
            for (int v = 0; v < 8; ++v) {
                int koff = 32 * kc + 2 * v + (v >= 4 ? 8 : 0) + 8 * half;
                aP.u[v] = *(const unsigned*)(sPw + l16 * 66 + koff);
            }
            #pragma unroll
            for (int ct = 0; ct < 4; ++ct) {
                FragH bV;
                const int crow = 16 * ct + l16;
                #pragma unroll
                for (int v = 0; v < 8; ++v) {
                    int soff = 32 * kc + 2 * v + (v >= 4 ? 8 : 0) + 8 * half;
                    bV.u[v] = *(const unsigned*)(sV + crow * 66 + soff);
                }
                Oacc[ct] = __builtin_amdgcn_wmma_f32_16x16x32_f16(
                             false, aP.v, false, bV.v, (short)0, Oacc[ct], false, false);
            }
        }
    }

    // ---- Normalize and stage output [c][t] in LDS, then coalesced HBM store ----
    __syncthreads();
    #pragma unroll
    for (int ct = 0; ct < 4; ++ct)
        #pragma unroll
        for (int r = 0; r < 8; ++r) {
            int c  = 16 * ct + l16;
            int tl = 16 * wave + r + 8 * half;
            sOut[c * 132 + tl] = Oacc[ct][r] / rSum[r];
        }
    __syncthreads();
    for (int i = tid; i < CH * T_TILE; i += 256) {
        int c = i >> 7, tl = i & 127;
        out[((size_t)bh * CH + c) * SEQ + t0 + tl] = sOut[c * 132 + tl];
    }
}

extern "C" void kernel_launch(void* const* d_in, const int* in_sizes, int n_in,
                              void* d_out, int out_size, void* d_ws, size_t ws_size,
                              hipStream_t stream) {
    (void)in_sizes; (void)n_in; (void)d_ws; (void)ws_size; (void)out_size;
    const float* qkv = (const float*)d_in[0];
    float* out = (float*)d_out;
    dim3 grid(SEQ / T_TILE, 4 * HEADS, 1);   // 16 t-tiles x 64 (b,head) pairs
    qkv_attn_kernel<<<grid, 256, 0, stream>>>(qkv, out);
}
